// BDHGPURefStabilized_14370960572596
// MI455X (gfx1250) — matrix operand order repaired and checked by reading, not compiled, not run on hardware
//
#include <hip/hip_runtime.h>
#include <hip/hip_bf16.h>

#define NN 2048
#define DD 128
#define TT 256
#define BB 8
#define NTHREADS 512
#define NWAVES 16
#define U_DECAY 0.97f
#define X_DECAY 0.97f
#define THR 0.02f

typedef __attribute__((ext_vector_type(16))) __bf16 v16bf;
typedef __attribute__((ext_vector_type(8)))  float  v8f;

__device__ __forceinline__ __bf16 f2bf(float f) {
    union { float f; unsigned u; } uf; uf.f = f;
    unsigned u = uf.u;
    unsigned short r = (unsigned short)((u + 0x7FFFu + ((u >> 16) & 1u)) >> 16);
    union { unsigned short s; __bf16 b; } ub; ub.s = r;
    return ub.b;
}

// xor-exchange across wave32 via ds_swizzle (group-of-32 mode, immediate pattern:
// offset = xor_mask<<10 | and_mask(0x1F)); no index math, no clamping.
template <int IMM>
__device__ __forceinline__ float swz_xor(float v) {
    return __int_as_float(__builtin_amdgcn_ds_swizzle(__float_as_int(v), IMM));
}
__device__ __forceinline__ float wave_sum(float v) {
    v += swz_xor<0x041F>(v);   // xor 1
    v += swz_xor<0x081F>(v);   // xor 2
    v += swz_xor<0x101F>(v);   // xor 4
    v += swz_xor<0x201F>(v);   // xor 8
    v += swz_xor<0x401F>(v);   // xor 16
    return v;
}
__device__ __forceinline__ float wave_max(float v) {
    v = fmaxf(v, swz_xor<0x041F>(v));
    v = fmaxf(v, swz_xor<0x081F>(v));
    v = fmaxf(v, swz_xor<0x101F>(v));
    v = fmaxf(v, swz_xor<0x201F>(v));
    v = fmaxf(v, swz_xor<0x401F>(v));
    return v;
}

// Per-lane K index for the A-matrix 16x32 layout (contiguous, matching B):
// lanes 0-15 : K0..15
// lanes 16-31: K16..31
__device__ __forceinline__ int kidx(int lane, int e) {
    return ((lane < 16) ? 0 : 16) + e;
}

// Prologue: convert fp32 weight matrix [rows x cols] (row-major) to bf16,
// pre-swizzled into per-lane WMMA A-tile order:
//   out[((tile*ksteps + s)*32 + lane)*16 + e] = W[tile*16 + (lane&15)][s*32 + kidx(lane,e)]
__global__ void swz_kernel(const float* __restrict__ W, __bf16* __restrict__ out,
                           int rows, int cols) {
    int idx = blockIdx.x * blockDim.x + threadIdx.x;
    int total = rows * cols;
    if (idx >= total) return;
    int e    = idx & 15;
    int lane = (idx >> 4) & 31;
    int rest = idx >> 9;               // tile*ksteps + s
    int ksteps = cols >> 5;
    int s    = rest % ksteps;
    int tile = rest / ksteps;
    int r = (tile << 4) + (lane & 15);
    int k = (s << 5) + kidx(lane, e);
    out[idx] = f2bf(W[r * cols + k]);
}

__global__ __launch_bounds__(NTHREADS)
void bdh_kernel(const float* __restrict__ token_emb,   // [V, 128]
                const int*   __restrict__ tokens,      // [8, 256]
                const __bf16* __restrict__ DxS,        // swizzled [2048x128]
                const __bf16* __restrict__ DyS,        // swizzled [2048x128]
                const __bf16* __restrict__ ES,         // swizzled [128x2048]
                float* __restrict__ rho_ws,            // [8][128][2048]
                float* __restrict__ out)               // [8][256][128]
{
    const int b    = blockIdx.x;
    const int tid  = threadIdx.x;
    const int lane = tid & 31;
    const int wave = tid >> 5;

    __shared__ __align__(16) float  x[NN];        // carried (post-threshold) state
    __shared__ __align__(16) float  xraw[NN];
    __shared__ __align__(16) float  vpbuf[2][DD]; // double-buffered v_prev (async-filled)
    __shared__ __align__(32) __bf16 vbf[DD];
    __shared__ __align__(32) __bf16 lnbf[DD];
    __shared__ __align__(32) __bf16 ytbf[NN];
    __shared__ __align__(16) float  astar[DD];
    __shared__ __align__(16) float  vpart[2][DD]; // split-K partials of E @ y
    __shared__ __align__(16) float  red[NWAVES];
    __shared__ __align__(16) float  redM[NWAVES];

    float* rho = rho_ws + (size_t)b * DD * NN;

    // zero rho (1 MB / WGP) and x; async-preload v_prev for t=0
    {
        float4 z4 = make_float4(0.f, 0.f, 0.f, 0.f);
        float4* rho4 = (float4*)rho;
        for (int i = tid; i < (DD * NN) / 4; i += NTHREADS) rho4[i] = z4;
        for (int i = tid; i < NN; i += NTHREADS) x[i] = 0.f;
        if (tid < 32) {
            int tok0 = tokens[b * TT];
            unsigned long long ga =
                (unsigned long long)(uintptr_t)(token_emb + (size_t)tok0 * DD) +
                (unsigned long long)lane * 16ull;
            unsigned la = (unsigned)(uintptr_t)&vpbuf[0][lane * 4];
            asm volatile("global_load_async_to_lds_b128 %0, %1, off"
                         :: "v"(la), "v"(ga) : "memory");
            asm volatile("s_wait_asynccnt 0" ::: "memory");
        }
    }
    __syncthreads();

    for (int t = 0; t < TT; ++t) {
        const int cur = t & 1;

        // ---- phase 0: convert v_prev to bf16; async-prefetch next token row ----
        if (wave == 0 && t + 1 < TT) {
            int tokn = tokens[b * TT + t + 1];
            unsigned long long ga =
                (unsigned long long)(uintptr_t)(token_emb + (size_t)tokn * DD) +
                (unsigned long long)lane * 16ull;
            unsigned la = (unsigned)(uintptr_t)&vpbuf[cur ^ 1][lane * 4];
            asm volatile("global_load_async_to_lds_b128 %0, %1, off"
                         :: "v"(la), "v"(ga) : "memory");
        }
        if (tid < DD) vbf[tid] = f2bf(vpbuf[cur][tid]);
        __syncthreads();                                              // B1

        // ---- phase 1: xraw = X_DECAY*x + Dx @ v_prev   (WMMA bf16) ----
        {
            const int khalf = (lane >= 16) ? 16 : 0;
            v16bf bmat[4];
            #pragma unroll
            for (int s = 0; s < 4; ++s)
                bmat[s] = *((const v16bf*)&vbf[s * 32 + khalf]);
            for (int tile = wave; tile < NN / 16; tile += NWAVES) {
                __builtin_prefetch(DxS + ((size_t)((tile + NWAVES) * 4) * 32 + lane) * 16, 0, 1);
                v8f c = {};
                #pragma unroll
                for (int s = 0; s < 4; ++s) {
                    v16bf a = *((const v16bf*)(DxS + ((size_t)(tile * 4 + s) * 32 + lane) * 16));
                    c = __builtin_amdgcn_wmma_f32_16x16x32_bf16(
                            false, a, false, bmat[s], (short)0, c, false, false);
                }
                if ((lane & 15) == 0) {               // column-0 holders: lanes 0 & 16
                    int rbase = tile * 16 + ((lane >= 16) ? 8 : 0);
                    #pragma unroll
                    for (int r = 0; r < 8; ++r)
                        xraw[rbase + r] = c[r] + X_DECAY * x[rbase + r];
                }
            }
        }
        __syncthreads();                                              // B2

        // ---- phase 2: L1-normalize, threshold at THR*max, write carried x ----
        {
            float ls = 0.f, lm = -3.4e38f;
            #pragma unroll
            for (int j = 0; j < 4; ++j) {
                float v = xraw[tid * 4 + j];
                ls += fabsf(v);
                lm  = fmaxf(lm, v);
            }
            ls = wave_sum(ls);
            lm = wave_max(lm);
            if (lane == 0) { red[wave] = ls; redM[wave] = lm; }
        }
        __syncthreads();                                              // B3
        {
            float rv = 0.f, rm = -3.4e38f;
            if (lane < NWAVES) { rv = red[lane]; rm = redM[lane]; }
            float tot = wave_sum(rv);
            float mx  = wave_max(rm);
            float inv = 1.0f / (tot + 1e-6f);
            float thr = THR * mx * inv;
            #pragma unroll
            for (int j = 0; j < 4; ++j) {
                int n = tid * 4 + j;
                float z = xraw[n] * inv;
                x[n] = (z > thr) ? z : 0.0f;
            }
        }
        __syncthreads();                                              // B4

        // ---- phase 3 (fused): a_star = rho_old @ x ; rho = U*(rho + v_prev (x) x) ----
        {
            int d  = tid >> 2;
            int cq = tid & 3;
            float vd  = vpbuf[cur][d];
            float acc = 0.f;
            float* rrow = rho + (size_t)d * NN;
            #pragma unroll 4
            for (int j = 0; j < NN / 16; ++j) {       // float4 strided: good coalescing
                int n4 = (j * 4 + cq) * 4;
                float4 rv = *((float4*)&rrow[n4]);
                float x0 = x[n4], x1 = x[n4 + 1], x2 = x[n4 + 2], x3 = x[n4 + 3];
                acc += rv.x * x0 + rv.y * x1 + rv.z * x2 + rv.w * x3;
                rv.x = U_DECAY * (rv.x + vd * x0);
                rv.y = U_DECAY * (rv.y + vd * x1);
                rv.z = U_DECAY * (rv.z + vd * x2);
                rv.w = U_DECAY * (rv.w + vd * x3);
                *((float4*)&rrow[n4]) = rv;
            }
            // 4 partials per d sit in adjacent lanes -> in-wave combine
            acc += swz_xor<0x041F>(acc);
            acc += swz_xor<0x081F>(acc);
            if (cq == 0) astar[d] = acc;
        }
        __syncthreads();                                              // B5

        // ---- phase 4: lnbf = LayerNorm(a_star), wave 0 only (ddof=1, eps on std) ----
        if (wave == 0) {
            float a0 = astar[lane * 4 + 0], a1 = astar[lane * 4 + 1];
            float a2 = astar[lane * 4 + 2], a3 = astar[lane * 4 + 3];
            float mean = wave_sum(a0 + a1 + a2 + a3) * (1.0f / DD);
            float d0 = a0 - mean, d1 = a1 - mean, d2 = a2 - mean, d3 = a3 - mean;
            float var  = wave_sum(d0 * d0 + d1 * d1 + d2 * d2 + d3 * d3) * (1.0f / (DD - 1));
            float rstd = 1.0f / (sqrtf(var) + 1e-6f);
            lnbf[lane * 4 + 0] = f2bf(d0 * rstd);
            lnbf[lane * 4 + 1] = f2bf(d1 * rstd);
            lnbf[lane * 4 + 2] = f2bf(d2 * rstd);
            lnbf[lane * 4 + 3] = f2bf(d3 * rstd);
        }
        __syncthreads();                                              // B6

        // ---- phase 5: y = relu(Dy @ ln_a) * relu(x)   (WMMA bf16) ----
        {
            const int khalf = (lane >= 16) ? 16 : 0;
            v16bf bmat[4];
            #pragma unroll
            for (int s = 0; s < 4; ++s)
                bmat[s] = *((const v16bf*)&lnbf[s * 32 + khalf]);
            for (int tile = wave; tile < NN / 16; tile += NWAVES) {
                __builtin_prefetch(DyS + ((size_t)((tile + NWAVES) * 4) * 32 + lane) * 16, 0, 1);
                v8f c = {};
                #pragma unroll
                for (int s = 0; s < 4; ++s) {
                    v16bf a = *((const v16bf*)(DyS + ((size_t)(tile * 4 + s) * 32 + lane) * 16));
                    c = __builtin_amdgcn_wmma_f32_16x16x32_bf16(
                            false, a, false, bmat[s], (short)0, c, false, false);
                }
                if ((lane & 15) == 0) {
                    int rbase = tile * 16 + ((lane >= 16) ? 8 : 0);
                    #pragma unroll
                    for (int r = 0; r < 8; ++r) {
                        int n = rbase + r;
                        ytbf[n] = f2bf(fmaxf(c[r], 0.f) * fmaxf(x[n], 0.f));
                    }
                }
            }
        }
        __syncthreads();                                              // B7

        // ---- phase 6: vpart[kg] = E[:, kg-half] @ y[kg-half]  (WMMA, split-K x2) ----
        {
            const int khalf = (lane >= 16) ? 16 : 0;
            const int tile  = wave >> 1;      // 0..7 : 16-row block of d
            const int kg    = wave & 1;       // K half
            v8f c = {};
            #pragma unroll 4
            for (int s = 0; s < 32; ++s) {
                int ks = kg * 32 + s;
                __builtin_prefetch(ES + ((size_t)(tile * 64 + ks + 4) * 32 + lane) * 16, 0, 1);
                v16bf a  = *((const v16bf*)(ES + ((size_t)(tile * 64 + ks) * 32 + lane) * 16));
                v16bf bm = *((const v16bf*)&ytbf[ks * 32 + khalf]);
                c = __builtin_amdgcn_wmma_f32_16x16x32_bf16(
                        false, a, false, bm, (short)0, c, false, false);
            }
            if ((lane & 15) == 0) {
                int rbase = tile * 16 + ((lane >= 16) ? 8 : 0);
                #pragma unroll
                for (int r = 0; r < 8; ++r) vpart[kg][rbase + r] = c[r];
            }
        }
        __syncthreads();                                              // B8

        // ---- phase 7: out[b,t,:] = LayerNorm(vpart[0]+vpart[1]), wave 0 only ----
        if (wave == 0) {
            float a0 = vpart[0][lane * 4 + 0] + vpart[1][lane * 4 + 0];
            float a1 = vpart[0][lane * 4 + 1] + vpart[1][lane * 4 + 1];
            float a2 = vpart[0][lane * 4 + 2] + vpart[1][lane * 4 + 2];
            float a3 = vpart[0][lane * 4 + 3] + vpart[1][lane * 4 + 3];
            float mean = wave_sum(a0 + a1 + a2 + a3) * (1.0f / DD);
            float d0 = a0 - mean, d1 = a1 - mean, d2 = a2 - mean, d3 = a3 - mean;
            float var  = wave_sum(d0 * d0 + d1 * d1 + d2 * d2 + d3 * d3) * (1.0f / (DD - 1));
            float rstd = 1.0f / (sqrtf(var) + 1e-6f);
            float* op = out + ((size_t)b * TT + t) * DD + lane * 4;
            op[0] = d0 * rstd; op[1] = d1 * rstd;
            op[2] = d2 * rstd; op[3] = d3 * rstd;
            // ensure next step's async v_prev has landed before the barrier
            asm volatile("s_wait_asynccnt 0" ::: "memory");
        }
        __syncthreads();                                              // B9
    }
}

extern "C" void kernel_launch(void* const* d_in, const int* in_sizes, int n_in,
                              void* d_out, int out_size, void* d_ws, size_t ws_size,
                              hipStream_t stream) {
    const float* E    = (const float*)d_in[0];   // [128, 2048]
    const float* Dx   = (const float*)d_in[1];   // [2048, 128]
    const float* Dy   = (const float*)d_in[2];   // [2048, 128]
    const float* temb = (const float*)d_in[3];   // [V, 128]
    const int*   toks = (const int*)d_in[4];     // [8, 256]

    char* ws = (char*)d_ws;
    __bf16* DxS = (__bf16*)ws;  ws += (size_t)NN * DD * sizeof(__bf16);
    __bf16* DyS = (__bf16*)ws;  ws += (size_t)NN * DD * sizeof(__bf16);
    __bf16* ES  = (__bf16*)ws;  ws += (size_t)DD * NN * sizeof(__bf16);
    float*  rho = (float*)ws;   // 8 MB, zeroed by main kernel every launch

    const int total = NN * DD;
    swz_kernel<<<(total + 255) / 256, 256, 0, stream>>>(Dx, DxS, NN, DD);
    swz_kernel<<<(total + 255) / 256, 256, 0, stream>>>(Dy, DyS, NN, DD);
    swz_kernel<<<(total + 255) / 256, 256, 0, stream>>>(E,  ES,  DD, NN);

    bdh_kernel<<<BB, NTHREADS, 0, stream>>>(temb, toks, DxS, DyS, ES, rho, (float*)d_out);
}
